// FactorizedSpectralConv_44753559224568
// MI455X (gfx1250) — matrix operand order repaired and checked by reading, not compiled, not run on hardware
//
#include <hip/hip_runtime.h>

// ---------------------------------------------------------------------------
// FactorizedSpectralConv for MI455X (gfx1250, wave32, WMMA bf16->f32).
//
// Pipeline (all bf16 intermediates, f32 accumulation in WMMA):
//   K_basis : build DFT basis F[720,384] and scaled inverse basis I[384,720]
//   K_prep  : repack pct/ipct into per-mode bf16 GEMM operands (zero padded)
//   K_zero  : zero the K-pad rows of XMt and col-pad of XM2
//   G1      : x[276480,720](f32) x F[720,384]       -> XMt  (mode-major)
//   G2      : 362x batched XMt_z[768,384] x pctT[m] -> coeffs[z][768][192]
//   K_cmul  : per-(c,l,m) complex multiply by weight (in place)
//   G3      : 362x batched coeffs_z x ipctT[m]      -> XM2[276480,384]
//   G4      : XM2 x I[384,720]                      -> out (f32, d_out)
//   K_copy  : residual = x -> d_out + 199065600
//
// A-tile staging in the bf16 GEMMs uses GLOBAL_LOAD_ASYNC_TO_LDS_B128
// (ASYNCcnt) when available; fragment reads use ds_load_b128 in the exact
// WMMA A/B lane layouts.  Workspace ~573 MB (layout in kernel_launch).
// ---------------------------------------------------------------------------

typedef __attribute__((ext_vector_type(16))) __bf16 v16bf;
typedef __attribute__((ext_vector_type(8)))  float  v8f;
typedef __attribute__((ext_vector_type(4)))  unsigned int u32x4;
typedef __attribute__((ext_vector_type(4)))  float  f32x4;
typedef __attribute__((ext_vector_type(4)))  int    i32x4;

#if defined(__has_builtin)
#if __has_builtin(__builtin_amdgcn_global_load_async_to_lds_b128)
#define HAS_ASYNC_B128 1
#endif
#if __has_builtin(__builtin_amdgcn_s_wait_asynccnt)
#define HAS_WAIT_ASYNC 1
#endif
#endif

__device__ __forceinline__ unsigned short f2bf(float f) {
  unsigned int u = __builtin_bit_cast(unsigned int, f);
  return (unsigned short)((u + 0x7fffu + ((u >> 16) & 1u)) >> 16);  // RNE
}
__device__ __forceinline__ float bf2f(unsigned short h) {
  unsigned int u = ((unsigned int)h) << 16;
  return __builtin_bit_cast(float, u);
}
__device__ __forceinline__ u32x4 ld16s(const unsigned short* p) {
  return *(const u32x4*)p;  // 16B aligned by construction -> ds_load_b128
}
__device__ __forceinline__ v16bf make_frag(u32x4 a, u32x4 b) {
  union { u32x4 u[2]; v16bf f; } t;
  t.u[0] = a; t.u[1] = b;
  return t.f;
}
__device__ __forceinline__ v8f zero8() {
  v8f r;
#pragma unroll
  for (int i = 0; i < 8; ++i) r[i] = 0.f;
  return r;
}

// 16B global -> LDS async copy (ASYNCcnt); falls back to ISA mnemonic.
__device__ __forceinline__ void async_ld_b128(const unsigned short* g,
                                              unsigned short* l) {
#if HAS_ASYNC_B128
  __builtin_amdgcn_global_load_async_to_lds_b128(
      (i32x4*)(void*)g,
      (__attribute__((address_space(3))) i32x4*)l, 0, 0);
#else
  unsigned loff = (unsigned)(unsigned long long)(
      __attribute__((address_space(3))) unsigned short*)l;
  asm volatile("global_load_async_to_lds_b128 %0, %1, off" ::"v"(loff), "v"(g)
               : "memory");
#endif
}
__device__ __forceinline__ void wait_async0() {
#if HAS_WAIT_ASYNC
  __builtin_amdgcn_s_wait_asynccnt(0);
#else
  asm volatile("s_wait_asynccnt 0x0" ::: "memory");
#endif
}

// --------------------------- basis generation ------------------------------
__global__ void k_basis(unsigned short* __restrict__ F,
                        unsigned short* __restrict__ I) {
  int idx = blockIdx.x * 256 + threadIdx.x;
  if (idx >= 720 * 384) return;
  // Forward DFT basis F[n][c], c = 2m+ri : Re -> cos, Im -> -sin (rfft sign).
  {
    int n = idx / 384, c = idx % 384;
    float v = 0.f;
    if (c < 362) {
      int m = c >> 1;
      int t = (m * n) % 720;  // exact integer angle reduction
      float a = (float)t * (6.2831853071795864f / 720.f);
      v = (c & 1) ? -__sinf(a) : __cosf(a);
    }
    F[idx] = f2bf(v);
  }
  // Inverse (irfft) basis I[k][n]: out[n] = X0/720 + (2/720) sum_m (Re cos - Im sin)
  {
    int k = idx / 720, n = idx % 720;
    float v = 0.f;
    if (k < 362) {
      int m = k >> 1;
      int t = (m * n) % 720;
      float a = (float)t * (6.2831853071795864f / 720.f);
      float s = (m == 0) ? (1.f / 720.f) : (2.f / 720.f);
      v = s * ((k & 1) ? -__sinf(a) : __cosf(a));
    }
    I[idx] = f2bf(v);
  }
}

// ------------------ repack Legendre bases per mode (bf16) ------------------
__global__ void k_prep(const float* __restrict__ pct,
                       const float* __restrict__ ipct,
                       unsigned short* __restrict__ pctT,
                       unsigned short* __restrict__ ipctT) {
  long idx = (long)blockIdx.x * 256 + threadIdx.x;
  const long TOT = 181L * 384 * 192;
  if (idx >= TOT) return;
  int m = (int)(idx / (384 * 192));
  int r = (int)(idx % (384 * 192));
  {  // pctT[m][h][l]  <- pct[h, l, m]  ([360,180,181])
    int h = r / 192, l = r % 192;
    float v = (h < 360 && l < 180) ? pct[((long)h * 180 + l) * 181 + m] : 0.f;
    pctT[idx] = f2bf(v);
  }
  {  // ipctT[m][l][h] <- ipct[h, l, m]
    int l = r / 384, h = r % 384;
    float v = (h < 360 && l < 180) ? ipct[((long)h * 180 + l) * 181 + m] : 0.f;
    ipctT[idx] = f2bf(v);
  }
}

// ----------------------- zero K-padding regions ----------------------------
__global__ void k_zero(unsigned short* __restrict__ XMt,
                       unsigned short* __restrict__ XM2) {
  long idx = (long)blockIdx.x * 256 + threadIdx.x;
  const long N1 = 384L * 768 * 24;   // XMt h-pad rows (h=360..383)
  if (idx < N1) {
    long col = idx / (768 * 24);
    long r = idx % (768 * 24);
    long bc = r / 24;
    long h = 360 + (r % 24);
    XMt[col * 294912 + bc * 384 + h] = 0;
  }
  const long N2 = 276480L * 22;      // XM2 col-pad (cols 362..383)
  if (idx < N2) {
    long row = idx / 22;
    long cp = 362 + (idx % 22);
    XM2[row * 384 + cp] = 0;
  }
}

// --------------- diagonal complex weight multiply (in place) ---------------
__global__ void k_cmul(unsigned short* __restrict__ coeffs,
                       const float* __restrict__ w) {
  long idx = (long)blockIdx.x * 256 + threadIdx.x;
  const long TOT = 181L * 768 * 180;
  if (idx >= TOT) return;
  int m = (int)(idx / (768 * 180));
  int r = (int)(idx % (768 * 180));
  int bc = r / 180, l = r % 180;
  int c = bc % 384;
  long pr = (long)(2 * m) * 147456 + (long)bc * 192 + l;
  long pi = pr + 147456;
  float ar = bf2f(coeffs[pr]);
  float ai = bf2f(coeffs[pi]);
  long wo = (((long)c * 180 + l) * 181 + m) * 2;
  float wr = w[wo], wi = w[wo + 1];
  coeffs[pr] = f2bf(ar * wr - ai * wi);
  coeffs[pi] = f2bf(ar * wi + ai * wr);
}

// ----------------------------- residual copy -------------------------------
__global__ void k_copy(const float* __restrict__ src, float* __restrict__ dst) {
  long i = (long)blockIdx.x * 256 + threadIdx.x;   // float4 index
  f32x4 v = *(const f32x4*)(src + i * 4);
  *(f32x4*)(dst + i * 4) = v;
}

// ---------------------------------------------------------------------------
// Generic WMMA GEMM: C[M,N] = A[M,K] * B[K,N], bf16 inputs (A optionally f32,
// converted during LDS staging), f32 accumulate. Block = 256 thr = 8 waves.
// Block tile 128x128, wave tile 32x64 (2x4 WMMA 16x16x32 fragments).
// AKIND: 0 = A is f32, 1 = A is bf16 (async LDS staging).  EPI = epilogue.
// ---------------------------------------------------------------------------
template <int AKIND, int EPI>
__global__ __launch_bounds__(256) void gemm_wmma(
    const void* __restrict__ Av, const unsigned short* __restrict__ Bv,
    void* __restrict__ Cv, int M, int N, int K, int lda, int ldb,
    long strideA, int ashift, long strideB, int bshift, long ep0, long ep1) {
  __shared__ __align__(16) unsigned short sA[128 * 40];  // [row][k] pad 40
  __shared__ __align__(16) unsigned short sB[128 * 40];  // [col][k] pad 40

  const int tid = threadIdx.x;
  const int lane = tid & 31;
  const int wave = tid >> 5;
  const int wm = wave & 3;   // wave row   (4 x 32 rows)
  const int wn = wave >> 2;  // wave col   (2 x 64 cols)
  const int z = blockIdx.z;
  const int m0 = blockIdx.y * 128;
  const int n0 = blockIdx.x * 128;

  const unsigned short* Bp = Bv + ((long)(z >> bshift)) * strideB;
  const long aoff = ((long)(z >> ashift)) * strideA;

  v8f acc[2][4];
#pragma unroll
  for (int i = 0; i < 2; ++i)
#pragma unroll
    for (int j = 0; j < 4; ++j) acc[i][j] = zero8();

  // staging assignments
  const int arow = tid >> 1;           // 0..127
  const int acol = (tid & 1) * 16;     // 0 / 16
  const int bn = tid & 127;            // 0..127
  const int bk = (tid >> 7) * 16;      // 0 / 16

  const int rlo = lane & 15;
  const int sel = lane >> 4;

  const int kTiles = (K + 31) / 32;
  for (int kt = 0; kt < kTiles; ++kt) {
    const int k0 = kt * 32;

    // ---- stage A tile (128 x 32) ----
    {
      const int grow = m0 + arow;
      if (AKIND == 0) {
        const float* A = (const float*)Av + aoff;
        const float* p = A + (long)grow * lda + k0 + acol;
        union { unsigned short s[16]; u32x4 v[2]; } pk;
        if (k0 + acol + 16 <= K) {
          if (kt + 1 < kTiles) __builtin_prefetch(p + 32, 0, 3);
#pragma unroll
          for (int i = 0; i < 4; ++i) {
            f32x4 fv = *(const f32x4*)(p + i * 4);
            pk.s[i * 4 + 0] = f2bf(fv.x);
            pk.s[i * 4 + 1] = f2bf(fv.y);
            pk.s[i * 4 + 2] = f2bf(fv.z);
            pk.s[i * 4 + 3] = f2bf(fv.w);
          }
        } else {
#pragma unroll
          for (int i = 0; i < 16; ++i) {
            int kk = k0 + acol + i;
            pk.s[i] = (kk < K) ? f2bf(A[(long)grow * lda + kk]) : 0;
          }
        }
        *(u32x4*)&sA[arow * 40 + acol] = pk.v[0];
        *(u32x4*)&sA[arow * 40 + acol + 8] = pk.v[1];
      } else {
        const unsigned short* A = (const unsigned short*)Av + aoff;
        const unsigned short* p = A + (long)grow * lda + k0 + acol;
        if (k0 + acol + 16 <= K) {
          // async VMEM -> LDS copy, tracked with ASYNCcnt
          async_ld_b128(p, &sA[arow * 40 + acol]);
          async_ld_b128(p + 8, &sA[arow * 40 + acol + 8]);
        } else {
          union { unsigned short s[16]; u32x4 v[2]; } pk;
#pragma unroll
          for (int i = 0; i < 16; ++i) {
            int kk = k0 + acol + i;
            pk.s[i] = (kk < K) ? A[(long)grow * lda + kk] : 0;
          }
          *(u32x4*)&sA[arow * 40 + acol] = pk.v[0];
          *(u32x4*)&sA[arow * 40 + acol + 8] = pk.v[1];
        }
      }
    }

    // ---- stage B tile transposed (sB[col][k], 128 x 32) ----
    {
      const int gn = n0 + bn;
      union { unsigned short s[16]; u32x4 v[2]; } pk;
#pragma unroll
      for (int i = 0; i < 16; ++i) {
        int kk = k0 + bk + i;
        pk.s[i] = (kk < K && gn < N) ? Bp[(long)kk * ldb + gn] : 0;
      }
      *(u32x4*)&sB[bn * 40 + bk] = pk.v[0];
      *(u32x4*)&sB[bn * 40 + bk + 8] = pk.v[1];
    }

    if (AKIND == 1) wait_async0();  // async LDS writes visible before barrier
    __syncthreads();

    // ---- fragments + 8 WMMAs ----
    v16bf af[2], bf[4];
#pragma unroll
    for (int fm = 0; fm < 2; ++fm) {
      const int row = wm * 32 + fm * 16 + rlo;
      const int kb = sel * 8;  // lanes 0-15: K 0-7/16-23, lanes 16-31: 8-15/24-31
      af[fm] = make_frag(ld16s(&sA[row * 40 + kb]), ld16s(&sA[row * 40 + kb + 16]));
    }
#pragma unroll
    for (int fn = 0; fn < 4; ++fn) {
      const int col = wn * 64 + fn * 16 + rlo;
      const int kb = sel * 8;  // lanes 0-15: K 0-7/16-23, lanes 16-31: 8-15/24-31
      bf[fn] = make_frag(ld16s(&sB[col * 40 + kb]), ld16s(&sB[col * 40 + kb + 16]));
    }
#pragma unroll
    for (int fm = 0; fm < 2; ++fm)
#pragma unroll
      for (int fn = 0; fn < 4; ++fn)
        acc[fm][fn] = __builtin_amdgcn_wmma_f32_16x16x32_bf16(
            false, af[fm], false, bf[fn], (short)0, acc[fm][fn], false, false);

    __syncthreads();
  }

  // ---- epilogue: C layout (f32 16x16): lane<16 -> (M=v, N=lane),
  //      lane>=16 -> (M=v+8, N=lane-16) ----
#pragma unroll
  for (int fm = 0; fm < 2; ++fm)
#pragma unroll
    for (int fn = 0; fn < 4; ++fn)
#pragma unroll
      for (int v = 0; v < 8; ++v) {
        const int gm = m0 + wm * 32 + fm * 16 + sel * 8 + v;
        const int gn = n0 + wn * 64 + fn * 16 + rlo;
        const float val = acc[fm][fn][v];
        if (EPI == 0) {  // G1 -> XMt[col=gn][bc*384 + h], ep0 = 294912
          if (gn < N) {
            const int bc = gm / 360;
            const int h = gm - bc * 360;
            ((unsigned short*)Cv)[(long)gn * ep0 + (long)bc * 384 + h] = f2bf(val);
          }
        } else if (EPI == 1) {  // G2 -> coeffs[z][gm][gn], ep0=147456 ep1=192
          if (gn < N)
            ((unsigned short*)Cv)[(long)z * ep0 + (long)gm * ep1 + gn] = f2bf(val);
        } else if (EPI == 2) {  // G3 -> XM2[(bc*360+h)][col=z], row stride 384
          if (gn < 360)
            ((unsigned short*)Cv)[((long)gm * 360 + gn) * 384 + z] = f2bf(val);
        } else {  // G4 -> out f32 [gm][gn], ep0 = 720
          if (gn < N) ((float*)Cv)[(long)gm * ep0 + gn] = val;
        }
      }
}

// ---------------------------------------------------------------------------
extern "C" void kernel_launch(void* const* d_in, const int* in_sizes, int n_in,
                              void* d_out, int out_size, void* d_ws,
                              size_t ws_size, hipStream_t stream) {
  (void)in_sizes; (void)n_in; (void)out_size; (void)ws_size;
  const float* x    = (const float*)d_in[0];
  const float* wgt  = (const float*)d_in[1];
  const float* pct  = (const float*)d_in[2];
  const float* ipct = (const float*)d_in[3];
  float* out = (float*)d_out;

  char* ws = (char*)d_ws;
  unsigned short* F      = (unsigned short*)(ws + 0L);
  unsigned short* Ibas   = (unsigned short*)(ws + 552960L);
  unsigned short* pctT   = (unsigned short*)(ws + 1105920L);
  unsigned short* ipctT  = (unsigned short*)(ws + 27795456L);
  unsigned short* XMt    = (unsigned short*)(ws + 54484992L);
  unsigned short* coeffs = (unsigned short*)(ws + 280977408L);
  unsigned short* XM2    = (unsigned short*)(ws + 387735552L);

  // bases + repack + pad-zeroing
  k_basis<<<1080, 256, 0, stream>>>(F, Ibas);
  k_prep<<<(int)((181L * 384 * 192 + 255) / 256), 256, 0, stream>>>(pct, ipct,
                                                                    pctT, ipctT);
  k_zero<<<(int)((384L * 768 * 24 + 255) / 256), 256, 0, stream>>>(XMt, XM2);

  // G1: forward DFT  (M=276480, N=384, K=720)
  gemm_wmma<0, 0><<<dim3(3, 2160, 1), 256, 0, stream>>>(
      x, F, XMt, 276480, 384, 720, 720, 384, 0L, 0, 0L, 0, 294912L, 0L);

  // G2: Legendre analysis, 362 batches (M=768, N=192, K=384)
  gemm_wmma<1, 1><<<dim3(2, 6, 362), 256, 0, stream>>>(
      XMt, pctT, coeffs, 768, 192, 384, 384, 192, 294912L, 0, 73728L, 1,
      147456L, 192L);

  // diagonal complex weight
  k_cmul<<<(int)((181L * 768 * 180 + 255) / 256), 256, 0, stream>>>(coeffs, wgt);

  // G3: Legendre synthesis, 362 batches (M=768, N=384, K=192)
  gemm_wmma<1, 2><<<dim3(3, 6, 362), 256, 0, stream>>>(
      coeffs, ipctT, XM2, 768, 384, 192, 192, 384, 147456L, 0, 73728L, 1, 0L,
      0L);

  // G4: inverse DFT  (M=276480, N=720, K=384) -> out (f32)
  gemm_wmma<1, 3><<<dim3(6, 2160, 1), 256, 0, stream>>>(
      XM2, Ibas, out, 276480, 720, 384, 384, 720, 0L, 0, 0L, 0, 720L, 0L);

  // residual passthrough
  k_copy<<<(int)(199065600L / 4 / 256), 256, 0, stream>>>(x, out + 199065600L);
}
